// Net_61220463837330
// MI455X (gfx1250) — compile-verified, matrix operand order checked by reading, not run on hardware
//
#include <hip/hip_runtime.h>
#include <math.h>

// ---------------------------------------------------------------------------
// S2VT video-captioning forward pass for MI455X (gfx1250, wave32, WMMA).
// Strategy: hoist all recurrence-independent GEMMs (feat@w_ih1^T,
// caption@w_ih2[:,:512]^T) into two large bf16 WMMA GEMMs; run the 109
// sequential LSTM steps as small WMMA GEMMs + fp32 elementwise cells.
// ---------------------------------------------------------------------------

typedef __attribute__((ext_vector_type(16))) __bf16 bf16x16;
typedef __attribute__((ext_vector_type(8)))  __bf16 bf16x8;
typedef __attribute__((ext_vector_type(4)))  __bf16 bf16x4;
typedef __attribute__((ext_vector_type(8)))  float  f32x8;

#define B_    128
#define TF_   80
#define DEC_  30
#define E_    512
#define FEAT_ 4096
#define V_    6000

__device__ __forceinline__ __bf16 f2bf(float x) {
  unsigned u = __builtin_bit_cast(unsigned, x);
  unsigned r = u + 0x7FFFu + ((u >> 16) & 1u);   // round-to-nearest-even
  unsigned short h = (unsigned short)(r >> 16);
  return __builtin_bit_cast(__bf16, h);
}

__device__ __forceinline__ bf16x16 load_frag(const __bf16* rowp, int khi) {
  // 16-bit A/B fragment layout (ISA 7.12.2): lane<16 holds K {0..7,16..23},
  // lane>=16 holds K {8..15,24..31}; two 16-byte LDS reads per fragment.
  bf16x8 lo = *(const bf16x8*)(rowp + 8 * khi);
  bf16x8 hi = *(const bf16x8*)(rowp + 16 + 8 * khi);
  bf16x16 v;
#pragma unroll
  for (int i = 0; i < 8; ++i) { v[i] = lo[i]; v[8 + i] = hi[i]; }
  return v;
}

// C[M,N] = A[M,K] @ Bw[N,K]^T (+ D[M,N]) (+ bias[N]); all row-major f32,
// staged to LDS as bf16, accumulated in f32 via v_wmma_f32_16x16x32_bf16.
// 64x64 tile per 128-thread block; 4 waves, each a 2x2 grid of 16x16 frags.
__global__ __launch_bounds__(128)
void gemm_wmma_bf16(const float* __restrict__ A, int lda,
                    const float* __restrict__ Bw, int ldb,
                    float* __restrict__ C, int ldc,
                    const float* __restrict__ D, int ldd,
                    const float* __restrict__ bias,
                    int M, int N, int K) {
  __shared__ __align__(16) __bf16 As[64][40];  // +8 bf16 pad: conflict-free b128
  __shared__ __align__(16) __bf16 Bs[64][40];

  const int tid  = threadIdx.x;
  const int lane = tid & 31;
  const int wv   = tid >> 5;          // wave 0..3
  const int wr   = (wv >> 1) * 32;    // wave row offset in 64x64 tile
  const int wc   = (wv & 1) * 32;     // wave col offset
  const int row0 = blockIdx.y * 64;
  const int col0 = blockIdx.x * 64;
  const int lm   = lane & 15;
  const int khi  = lane >> 4;

  f32x8 acc[2][2] = {};

  for (int k0 = 0; k0 < K; k0 += 32) {
    // Cooperative stage: 64x32 f32 -> bf16 per matrix; 4 floats/thread/pass.
#pragma unroll
    for (int p = 0; p < 4; ++p) {
      int e  = (p * 128 + tid) * 4;
      int r  = e >> 5;
      int cc = e & 31;
      float4 va = make_float4(0.f, 0.f, 0.f, 0.f);
      int gr = row0 + r;
      if (gr < M) va = *(const float4*)(A + (size_t)gr * lda + k0 + cc);
      bf16x4 sa; sa[0] = f2bf(va.x); sa[1] = f2bf(va.y);
      sa[2] = f2bf(va.z); sa[3] = f2bf(va.w);
      *(bf16x4*)&As[r][cc] = sa;

      float4 vb = make_float4(0.f, 0.f, 0.f, 0.f);
      int gn = col0 + r;
      if (gn < N) vb = *(const float4*)(Bw + (size_t)gn * ldb + k0 + cc);
      bf16x4 sb; sb[0] = f2bf(vb.x); sb[1] = f2bf(vb.y);
      sb[2] = f2bf(vb.z); sb[3] = f2bf(vb.w);
      *(bf16x4*)&Bs[r][cc] = sb;
    }
    __syncthreads();

    bf16x16 a0 = load_frag(&As[wr + lm][0], khi);
    bf16x16 a1 = load_frag(&As[wr + 16 + lm][0], khi);
    bf16x16 b0 = load_frag(&Bs[wc + lm][0], khi);
    bf16x16 b1 = load_frag(&Bs[wc + 16 + lm][0], khi);

    acc[0][0] = __builtin_amdgcn_wmma_f32_16x16x32_bf16(
        false, a0, false, b0, (short)0, acc[0][0], false, false);
    acc[0][1] = __builtin_amdgcn_wmma_f32_16x16x32_bf16(
        false, a0, false, b1, (short)0, acc[0][1], false, false);
    acc[1][0] = __builtin_amdgcn_wmma_f32_16x16x32_bf16(
        false, a1, false, b0, (short)0, acc[1][0], false, false);
    acc[1][1] = __builtin_amdgcn_wmma_f32_16x16x32_bf16(
        false, a1, false, b1, (short)0, acc[1][1], false, false);
    __syncthreads();
  }

  // C/D f32 layout: VGPR r -> M = r + 8*(lane>=16), N = lane&15.
#pragma unroll
  for (int mi = 0; mi < 2; ++mi)
#pragma unroll
    for (int ni = 0; ni < 2; ++ni)
#pragma unroll
      for (int r = 0; r < 8; ++r) {
        int m = row0 + wr + mi * 16 + 8 * khi + r;
        int n = col0 + wc + ni * 16 + lm;
        if (m < M && n < N) {
          float v = acc[mi][ni][r];
          if (D)    v += D[(size_t)m * ldd + n];
          if (bias) v += bias[n];
          C[(size_t)m * ldc + n] = v;
        }
      }
}

// PyTorch LSTMCell gate order (i,f,g,o); gates [128,2048], h/c [128,512].
__global__ void lstm_cell_kernel(const float* __restrict__ gates,
                                 float* __restrict__ h, float* __restrict__ c) {
  int idx = blockIdx.x * blockDim.x + threadIdx.x;   // 0..65535
  int b = idx >> 9, j = idx & 511;
  const float* g = gates + (size_t)b * 2048;
  float ig = 1.f / (1.f + expf(-g[j]));
  float fg = 1.f / (1.f + expf(-g[512 + j]));
  float gg = tanhf(g[1024 + j]);
  float og = 1.f / (1.f + expf(-g[1536 + j]));
  float cn = fg * c[idx] + ig * gg;
  c[idx] = cn;
  h[idx] = og * tanhf(cn);
}

__global__ void argmax_kernel(const float* __restrict__ oh,
                              int* __restrict__ labels) {
  int row = blockIdx.x;                     // b*DEC + t
  const float* p = oh + (size_t)row * V_;
  float bv = -1e30f; int bi = 0;
  for (int i = threadIdx.x; i < V_; i += 256) {
    float v = p[i];
    if (v > bv) { bv = v; bi = i; }
  }
  __shared__ float sv[256];
  __shared__ int   si[256];
  sv[threadIdx.x] = bv; si[threadIdx.x] = bi;
  __syncthreads();
  for (int s = 128; s > 0; s >>= 1) {
    if (threadIdx.x < s) {
      float v2 = sv[threadIdx.x + s]; int i2 = si[threadIdx.x + s];
      if (v2 > sv[threadIdx.x] ||
          (v2 == sv[threadIdx.x] && i2 < si[threadIdx.x])) {
        sv[threadIdx.x] = v2; si[threadIdx.x] = i2;
      }
    }
    __syncthreads();
  }
  if (threadIdx.x == 0) labels[row] = si[0];
}

__global__ void loss_kernel(const float* __restrict__ logits,
                            const int* __restrict__ labels, int t1,
                            float* __restrict__ out) {
  int b = blockIdx.x;
  const float* p = logits + (size_t)b * V_;
  __shared__ float red[256];
  float mx = -1e30f;
  for (int i = threadIdx.x; i < V_; i += 256) mx = fmaxf(mx, p[i]);
  red[threadIdx.x] = mx; __syncthreads();
  for (int s = 128; s > 0; s >>= 1) {
    if (threadIdx.x < s)
      red[threadIdx.x] = fmaxf(red[threadIdx.x], red[threadIdx.x + s]);
    __syncthreads();
  }
  mx = red[0]; __syncthreads();
  float sum = 0.f;
  for (int i = threadIdx.x; i < V_; i += 256) sum += expf(p[i] - mx);
  red[threadIdx.x] = sum; __syncthreads();
  for (int s = 128; s > 0; s >>= 1) {
    if (threadIdx.x < s) red[threadIdx.x] += red[threadIdx.x + s];
    __syncthreads();
  }
  if (threadIdx.x == 0) {
    int lab = labels[b * DEC_ + t1];
    float logp = p[lab] - mx - logf(red[0]);
    atomicAdd(out, -logp / (float)B_);
  }
}

__global__ void init_kernel(float* __restrict__ state, int n,
                            float* __restrict__ out) {
  int i = blockIdx.x * blockDim.x + threadIdx.x;
  if (i < n) state[i] = 0.f;
  if (i == 0) out[0] = 0.f;
}

__global__ void bias_kernel(const float* __restrict__ a1,
                            const float* __restrict__ a2,
                            float* __restrict__ o1,
                            const float* __restrict__ b1,
                            const float* __restrict__ b2,
                            float* __restrict__ o2) {
  int i = blockIdx.x * blockDim.x + threadIdx.x;
  if (i < 2048) { o1[i] = a1[i] + a2[i]; o2[i] = b1[i] + b2[i]; }
}

extern "C" void kernel_launch(void* const* d_in, const int* in_sizes, int n_in,
                              void* d_out, int out_size, void* d_ws,
                              size_t ws_size, hipStream_t stream) {
  (void)in_sizes; (void)n_in; (void)out_size; (void)ws_size;
  const float* feat    = (const float*)d_in[0];
  const float* caption = (const float*)d_in[1];
  const float* onehot  = (const float*)d_in[2];
  const float* w_ih1   = (const float*)d_in[3];
  const float* w_hh1   = (const float*)d_in[4];
  const float* b_ih1   = (const float*)d_in[5];
  const float* b_hh1   = (const float*)d_in[6];
  const float* w_ih2   = (const float*)d_in[7];
  const float* w_hh2   = (const float*)d_in[8];
  const float* b_ih2   = (const float*)d_in[9];
  const float* b_hh2   = (const float*)d_in[10];
  const float* w_out   = (const float*)d_in[11];
  const float* b_out   = (const float*)d_in[12];
  float* out = (float*)d_out;

  // Workspace carve-up (~122 MB of f32).
  float* p      = (float*)d_ws;
  float* Xenc   = p; p += (size_t)B_ * TF_ * 2048;   // feat @ w_ih1^T
  float* Xdec   = p; p += (size_t)B_ * DEC_ * 2048;  // caption @ w_ih2[:,:E]^T
  float* g1     = p; p += (size_t)B_ * 2048;
  float* g2     = p; p += (size_t)B_ * 2048;
  float* h1     = p; p += (size_t)B_ * E_;
  float* c1     = p; p += (size_t)B_ * E_;
  float* h2     = p; p += (size_t)B_ * E_;
  float* c2     = p; p += (size_t)B_ * E_;
  float* logits = p; p += (size_t)B_ * V_;
  float* b1c    = p; p += 2048;
  float* b2c    = p; p += 2048;
  int*   labels = (int*)p;                           // B_*DEC_ ints

  init_kernel<<<(4 * B_ * E_ + 255) / 256, 256, 0, stream>>>(h1, 4 * B_ * E_, out);
  bias_kernel<<<8, 256, 0, stream>>>(b_ih1, b_hh1, b1c, b_ih2, b_hh2, b2c);
  argmax_kernel<<<B_ * DEC_, 256, 0, stream>>>(onehot, labels);

  // Hoisted input GEMMs (recurrence-independent): the 172-GFLOP bulk.
  gemm_wmma_bf16<<<dim3(2048 / 64, (B_ * TF_) / 64), 128, 0, stream>>>(
      feat, FEAT_, w_ih1, FEAT_, Xenc, 2048, nullptr, 0, nullptr,
      B_ * TF_, 2048, FEAT_);
  gemm_wmma_bf16<<<dim3(2048 / 64, (B_ * DEC_) / 64), 128, 0, stream>>>(
      caption, E_, w_ih2, 2 * E_, Xdec, 2048, nullptr, 0, nullptr,
      B_ * DEC_, 2048, E_);

  dim3 gstep(2048 / 64, B_ / 64);
  // Encoder: LSTM1 input = feat (precomputed); LSTM2 input = [0, h1].
  for (int t = 0; t < TF_; ++t) {
    gemm_wmma_bf16<<<gstep, 128, 0, stream>>>(
        h1, E_, w_hh1, E_, g1, 2048,
        Xenc + (size_t)t * 2048, TF_ * 2048, b1c, B_, 2048, E_);
    lstm_cell_kernel<<<256, 256, 0, stream>>>(g1, h1, c1);
    gemm_wmma_bf16<<<gstep, 128, 0, stream>>>(
        h1, E_, w_ih2 + E_, 2 * E_, g2, 2048, nullptr, 0, b2c, B_, 2048, E_);
    gemm_wmma_bf16<<<gstep, 128, 0, stream>>>(
        h2, E_, w_hh2, E_, g2, 2048, g2, 2048, nullptr, B_, 2048, E_);
    lstm_cell_kernel<<<256, 256, 0, stream>>>(g2, h2, c2);
  }

  // Decoder: LSTM1 input = 0 (bias only); LSTM2 input = [word, h1].
  dim3 gout((V_ + 63) / 64, B_ / 64);
  for (int t = 0; t < DEC_ - 1; ++t) {
    gemm_wmma_bf16<<<gstep, 128, 0, stream>>>(
        h1, E_, w_hh1, E_, g1, 2048, nullptr, 0, b1c, B_, 2048, E_);
    lstm_cell_kernel<<<256, 256, 0, stream>>>(g1, h1, c1);
    gemm_wmma_bf16<<<gstep, 128, 0, stream>>>(
        h1, E_, w_ih2 + E_, 2 * E_, g2, 2048,
        Xdec + (size_t)t * 2048, DEC_ * 2048, b2c, B_, 2048, E_);
    gemm_wmma_bf16<<<gstep, 128, 0, stream>>>(
        h2, E_, w_hh2, E_, g2, 2048, g2, 2048, nullptr, B_, 2048, E_);
    lstm_cell_kernel<<<256, 256, 0, stream>>>(g2, h2, c2);
    gemm_wmma_bf16<<<gout, 128, 0, stream>>>(
        h2, E_, w_out, E_, logits, V_, nullptr, 0, b_out, B_, V_, E_);
    loss_kernel<<<B_, 256, 0, stream>>>(logits, labels, t + 1, out);
  }
}
